// Llama2Attention_34385508172327
// MI455X (gfx1250) — compile-verified, matrix operand order checked by reading, not run on hardware
//
#include <hip/hip_runtime.h>
#include <hip/hip_bf16.h>
#include <cstdint>
#include <cstddef>

typedef __attribute__((ext_vector_type(16))) __bf16 v16bf;
typedef __attribute__((ext_vector_type(8)))  __bf16 v8bf;
typedef __attribute__((ext_vector_type(8)))  float  v8f;
typedef __attribute__((ext_vector_type(4)))  float  v4f;
typedef __attribute__((ext_vector_type(4)))  int    v4i;
typedef __attribute__((ext_vector_type(4)))  uint32_t su4;
typedef __attribute__((ext_vector_type(8)))  uint32_t su8;

#define SEQ     2048
#define PAST    1024
#define TOTAL   3072
#define NSTATE  4096
#define NHEAD   32
#define HDIM    128

#ifndef USE_TDM
#define USE_TDM 1
#endif

// ---------------------------------------------------------------------------
// CDNA5 async global->LDS copy (per-lane, ASYNCcnt) with sync fallback.
// Builtin signature (from hipcc diagnostic): param0 = v4i __device__* (as1),
// param1 = v4i lds pointer (as3), then imm offset + imm cpol.
// as1 global pointers share the generic 64-bit value; the as3 LDS offset is
// the low 32 bits of the generic pointer (ISA 10.2 aperture mapping), so
// integer-laundered casts are semantically correct.
// ---------------------------------------------------------------------------
#if __has_builtin(__builtin_amdgcn_global_load_async_to_lds_b128)
#define HAVE_ASYNC_LDS 1
typedef __attribute__((address_space(1))) v4i as1_v4i;
typedef __attribute__((address_space(3))) v4i as3_v4i;
__device__ __forceinline__ void cp_async16(void* lds, const void* glb) {
  __builtin_amdgcn_global_load_async_to_lds_b128(
      (as1_v4i*)(uintptr_t)glb, (as3_v4i*)(uint32_t)(uintptr_t)lds, 0, 0);
}
#if __has_builtin(__builtin_amdgcn_s_wait_asynccnt)
#define WAIT_ASYNC() __builtin_amdgcn_s_wait_asynccnt(0)
#else
#define WAIT_ASYNC() asm volatile("s_wait_asynccnt 0x0" ::: "memory")
#endif
#endif

#if USE_TDM
// Tensor Data Mover: 2D tile load, 16-bit elements, LDS row padding
// 256B data + 16B pad per row (matches [][136] __bf16 arrays).
// D# group0/group1 packing per CDNA5 ISA.
__device__ __forceinline__ void tdm_load_tile_2d(uint32_t lds_addr,
                                                 const void* gaddr,
                                                 uint32_t tile_w,      // elems
                                                 uint32_t tile_h,      // rows
                                                 uint32_t stride_e,    // elems
                                                 uint32_t tensor_w,    // elems
                                                 uint32_t tensor_h) {  // rows
  uint64_t ga = (uint64_t)(uintptr_t)gaddr;
  su4 g0;
  g0[0] = 1u;                                   // count=1, user descriptor
  g0[1] = lds_addr;                             // lds_addr (bytes)
  g0[2] = (uint32_t)ga;                         // global_addr[31:0]
  g0[3] = (uint32_t)((ga >> 32) & 0x01FFFFFFu)  // global_addr[56:32]
        | (2u << 30);                           // type = 2 ("image")
  su8 g1;
  g1[0] = (1u << 16)                            // data_size = 2 bytes
        | (1u << 20)                            // pad_enable
        | (5u << 22)                            // pad_interval: 64 DWORDs
        | (3u << 25);                           // pad_amount: 4 DWORDs
  g1[1] = (tensor_w & 0xFFFFu) << 16;           // tensor_dim0[15:0]
  g1[2] = (tensor_w >> 16) | ((tensor_h & 0xFFFFu) << 16);  // dim0 hi | dim1 lo
  g1[3] = (tensor_h >> 16) | (tile_w << 16);    // dim1 hi | tile_dim0
  g1[4] = tile_h;                               // tile_dim1 (tile_dim2 = 0)
  g1[5] = stride_e;                             // tensor_dim0_stride[31:0]
  g1[6] = 0;                                    // stride hi | dim1_stride lo
  g1[7] = 0;
  asm volatile("tensor_load_to_lds %0, %1" :: "s"(g0), "s"(g1) : "memory");
}
__device__ __forceinline__ void tdm_wait() {
#if __has_builtin(__builtin_amdgcn_s_wait_tensorcnt)
  __builtin_amdgcn_s_wait_tensorcnt(0);
#else
  asm volatile("s_wait_tensorcnt 0x0" ::: "memory");
#endif
}
#endif

// ---------------------------------------------------------------------------
// WMMA fragment helpers (CDNA5 16x16x32 bf16, wave32)
// ---------------------------------------------------------------------------
__device__ __forceinline__ v16bf ldfrag(const __bf16* p) {
  v8bf lo = *(const v8bf*)(p);
  v8bf hi = *(const v8bf*)(p + 16);
  v16bf o;
#pragma unroll
  for (int i = 0; i < 8; ++i) { o[i] = lo[i]; o[i + 8] = hi[i]; }
  return o;
}

__device__ __forceinline__ v8f wmma_bf16(v16bf a, v16bf b, v8f c) {
  return __builtin_amdgcn_wmma_f32_16x16x32_bf16(false, a, false, b,
                                                 (short)0, c, false, false);
}

// ---------------------------------------------------------------------------
// fp32 -> bf16 conversion
// ---------------------------------------------------------------------------
__global__ void k_cvt_bf16(const float* __restrict__ src,
                           __bf16* __restrict__ dst, int n) {
  int i = (blockIdx.x * blockDim.x + threadIdx.x) * 4;
  if (i + 3 < n) {
    v4f v = *(const v4f*)(src + i);
    dst[i + 0] = (__bf16)v.x;
    dst[i + 1] = (__bf16)v.y;
    dst[i + 2] = (__bf16)v.z;
    dst[i + 3] = (__bf16)v.w;
  }
}

// ---------------------------------------------------------------------------
// NT GEMM:  C[M x N] = A[M x K](bf16) * B[N x K](fp32)^T
// 128x128 tile / 256 threads (8 waves), wave tile 32x64, K-step 32.
// Double-buffered LDS; A staged with async global->LDS (ASYNCcnt),
// B converted fp32->bf16 while staging.  One barrier per K-step.
// ---------------------------------------------------------------------------
template <bool BF16OUT>
__global__ __launch_bounds__(256) void k_gemm_nt(const __bf16* __restrict__ A,
                                                 const float* __restrict__ B,
                                                 void* __restrict__ Cout,
                                                 int K, int ldc) {
  __shared__ __bf16 As[2][128][40];
  __shared__ __bf16 Bs[2][128][40];

  const int tid  = threadIdx.x;
  const int lane = tid & 31;
  const int w    = tid >> 5;
  const int wm   = w >> 1;         // 0..3
  const int wn   = w & 1;          // 0..1
  const int m0   = blockIdx.y * 128;
  const int n0   = blockIdx.x * 128;

  const int r    = lane & 15;
  const int koff = (lane >> 4) << 3;

  v8f acc[2][4];
#pragma unroll
  for (int i = 0; i < 2; ++i)
#pragma unroll
    for (int j = 0; j < 4; ++j) acc[i][j] = {};

  const int arow = tid >> 1;          // 0..127
  const int acol = (tid & 1) * 16;    // 0 or 16

  auto stage = [&](int buf, int k0) {
    const __bf16* ap = A + (size_t)(m0 + arow) * K + k0 + acol;
#if defined(HAVE_ASYNC_LDS)
    cp_async16(&As[buf][arow][acol], ap);
    cp_async16(&As[buf][arow][acol + 8], ap + 8);
#else
    v8bf a0 = *(const v8bf*)(ap);
    v8bf a1 = *(const v8bf*)(ap + 8);
    *(v8bf*)(&As[buf][arow][acol])     = a0;
    *(v8bf*)(&As[buf][arow][acol + 8]) = a1;
#endif
    const float* bp = B + (size_t)(n0 + arow) * K + k0 + acol;
    v4f b0 = *(const v4f*)(bp);
    v4f b1 = *(const v4f*)(bp + 4);
    v4f b2 = *(const v4f*)(bp + 8);
    v4f b3 = *(const v4f*)(bp + 12);
    v8bf bb0, bb1;
#pragma unroll
    for (int e = 0; e < 4; ++e) {
      bb0[e]     = (__bf16)b0[e];
      bb0[e + 4] = (__bf16)b1[e];
      bb1[e]     = (__bf16)b2[e];
      bb1[e + 4] = (__bf16)b3[e];
    }
    *(v8bf*)(&Bs[buf][arow][acol])     = bb0;
    *(v8bf*)(&Bs[buf][arow][acol + 8]) = bb1;
    __builtin_prefetch(bp + 32, 0, 0);   // next B tile -> global_prefetch_b8
  };

  stage(0, 0);
  const int KT = K >> 5;
#pragma unroll 2
  for (int kt = 0; kt < KT; ++kt) {
    const int cur = kt & 1;
#if defined(HAVE_ASYNC_LDS)
    WAIT_ASYNC();                 // my async copies done (cross-wave via barrier)
#endif
    __syncthreads();              // staging visible; prev-iter reads retired
    if (kt + 1 < KT) stage(cur ^ 1, (kt + 1) << 5);

    v16bf af[2], bfr[4];
#pragma unroll
    for (int i = 0; i < 2; ++i)
      af[i] = ldfrag(&As[cur][wm * 32 + i * 16 + r][koff]);
#pragma unroll
    for (int j = 0; j < 4; ++j)
      bfr[j] = ldfrag(&Bs[cur][wn * 64 + j * 16 + r][koff]);

#pragma unroll
    for (int i = 0; i < 2; ++i)
#pragma unroll
      for (int j = 0; j < 4; ++j)
        acc[i][j] = wmma_bf16(af[i], bfr[j], acc[i][j]);
  }

  const int mg = (lane >> 4) * 8;
#pragma unroll
  for (int i = 0; i < 2; ++i)
#pragma unroll
    for (int j = 0; j < 4; ++j)
#pragma unroll
      for (int vr = 0; vr < 8; ++vr) {
        int row = m0 + wm * 32 + i * 16 + mg + vr;
        int col = n0 + wn * 64 + j * 16 + r;
        float val = acc[i][j][vr];
        if (BF16OUT)
          ((__bf16*)Cout)[(size_t)row * ldc + col] = (__bf16)val;
        else
          ((float*)Cout)[(size_t)row * ldc + col] = val;
      }
}

// ---------------------------------------------------------------------------
// RoPE (interleaved pairs), cache copies, V output
// ---------------------------------------------------------------------------
__global__ void k_rope_q(__bf16* __restrict__ q,
                         const float* __restrict__ cosT,
                         const float* __restrict__ sinT) {
  int idx = blockIdx.x * blockDim.x + threadIdx.x;
  if (idx >= SEQ * NHEAD * 64) return;
  int i  = idx & 63;
  int hh = (idx >> 6) & 31;
  int s  = idx >> 11;
  float c  = cosT[s * 64 + i];
  float sn = sinT[s * 64 + i];
  size_t base = (size_t)s * NSTATE + hh * HDIM + 2 * i;
  float t0 = (float)q[base], t1 = (float)q[base + 1];
  q[base]     = (__bf16)(t0 * c - t1 * sn);
  q[base + 1] = (__bf16)(t0 * sn + t1 * c);
}

__global__ void k_rope_k(__bf16* __restrict__ kb, float* __restrict__ kout,
                         const float* __restrict__ cosT,
                         const float* __restrict__ sinT) {
  int idx = blockIdx.x * blockDim.x + threadIdx.x;
  if (idx >= SEQ * NHEAD * 64) return;
  int i  = idx & 63;
  int hh = (idx >> 6) & 31;
  int s  = idx >> 11;
  float c  = cosT[s * 64 + i];
  float sn = sinT[s * 64 + i];
  size_t base = (size_t)(PAST + s) * NSTATE + hh * HDIM + 2 * i;
  float t0 = (float)kb[base], t1 = (float)kb[base + 1];
  float o0 = t0 * c - t1 * sn;
  float o1 = t0 * sn + t1 * c;
  kb[base]       = (__bf16)o0;
  kb[base + 1]   = (__bf16)o1;
  kout[base]     = o0;
  kout[base + 1] = o1;
}

__global__ void k_cache_store(const float* __restrict__ cache,
                              __bf16* __restrict__ bws,
                              float* __restrict__ outp) {
  int i = (blockIdx.x * blockDim.x + threadIdx.x) * 4;
  if (i + 3 < PAST * NSTATE) {
    v4f v = *(const v4f*)(cache + i);
#pragma unroll
    for (int e = 0; e < 4; ++e) {
      outp[i + e] = v[e];
      bws[i + e]  = (__bf16)v[e];
    }
  }
}

__global__ void k_vnew_store(const __bf16* __restrict__ vb,
                             float* __restrict__ vout) {
  int i = (blockIdx.x * blockDim.x + threadIdx.x) * 4;
  if (i + 3 < SEQ * NSTATE) {
    size_t base = (size_t)PAST * NSTATE + i;
#pragma unroll
    for (int e = 0; e < 4; ++e) vout[base + e] = (float)vb[base + e];
  }
}

// ---------------------------------------------------------------------------
// Flash attention: WG = (head, 64 q rows), 4 waves x 16 rows.
// K tile staged by the Tensor Data Mover (wave 0, TENSORcnt); V staged
// transposed by all threads; online softmax via per-wave LDS scoreboard.
// ---------------------------------------------------------------------------
__global__ __launch_bounds__(128) void k_attn(const __bf16* __restrict__ Q,
                                              const __bf16* __restrict__ Kb,
                                              const __bf16* __restrict__ Vb,
                                              __bf16* __restrict__ Ctx) {
  const int h    = blockIdx.y;
  const int q0   = blockIdx.x * 64;
  const int tid  = threadIdx.x;
  const int lane = tid & 31;
  const int w    = tid >> 5;
  const int r    = lane & 15;
  const int koff = (lane >> 4) << 3;
  const int mg   = (lane >> 4) * 8;

  __shared__ __bf16 Ks[64][136];             // keys x d (136 = 128 + TDM pad)
  __shared__ __bf16 Vt[128][72];             // d x keys (transposed)
  __shared__ float  Sl[4][16][64];
  __shared__ __bf16 Pl[4][16][72];
  __shared__ float  Ml[4][16], Ll[4][16], Al[4][16];

  v16bf qf[4];
  const int qrow = q0 + w * 16 + r;
#pragma unroll
  for (int kk = 0; kk < 4; ++kk)
    qf[kk] = ldfrag(Q + (size_t)qrow * NSTATE + h * HDIM + kk * 32 + koff);

  v8f o[8];
#pragma unroll
  for (int j = 0; j < 8; ++j) o[j] = {};
  if (lane < 16) { Ml[w][lane] = -1e30f; Ll[w][lane] = 0.0f; }

  const float scale  = 0.088388347648318447f;   // 1/sqrt(128)
  const int   ntiles = (PAST + q0 + 64) >> 6;

  for (int t = 0; t < ntiles; ++t) {
    const int j0 = t * 64;
    __syncthreads();   // previous-iteration readers done

#if USE_TDM
    if (w == 0)
      tdm_load_tile_2d((uint32_t)(uintptr_t)&Ks[0][0],
                       Kb + (size_t)j0 * NSTATE + h * HDIM,
                       HDIM, 64, NSTATE, NSTATE, TOTAL);
#endif
    {
      int krow = tid >> 1;
      int kc   = (tid & 1) * 64;
#if !USE_TDM
      const __bf16* kp = Kb + (size_t)(j0 + krow) * NSTATE + h * HDIM + kc;
#pragma unroll
      for (int u = 0; u < 8; ++u)
        *(v8bf*)(&Ks[krow][kc + u * 8]) = *(const v8bf*)(kp + u * 8);
#endif
      const __bf16* vp = Vb + (size_t)(j0 + krow) * NSTATE + h * HDIM + kc;
#pragma unroll
      for (int u = 0; u < 8; ++u) {
        v8bf vv = *(const v8bf*)(vp + u * 8);
#pragma unroll
        for (int e = 0; e < 8; ++e) Vt[kc + u * 8 + e][krow] = vv[e];
      }
    }
#if USE_TDM
    if (w == 0) tdm_wait();
#endif
    __syncthreads();

    // S = Q * K^T
    v8f sc[4];
#pragma unroll
    for (int nt = 0; nt < 4; ++nt) {
      v8f c = {};
#pragma unroll
      for (int kk = 0; kk < 4; ++kk) {
        v16bf bfrag = ldfrag(&Ks[nt * 16 + r][kk * 32 + koff]);
        c = wmma_bf16(qf[kk], bfrag, c);
      }
      sc[nt] = c;
    }

    // scale + causal mask
#pragma unroll
    for (int nt = 0; nt < 4; ++nt) {
      int kpos = j0 + nt * 16 + r;
#pragma unroll
      for (int vr = 0; vr < 8; ++vr) {
        int m    = mg + vr;
        int qpos = PAST + q0 + w * 16 + m;
        float v  = (kpos <= qpos) ? sc[nt][vr] * scale : -1e30f;
        Sl[w][m][nt * 16 + r] = v;
      }
    }
    __syncthreads();

    // online softmax (lanes 0..15 own one row each)
    if (lane < 16) {
      float mOld = Ml[w][lane];
      float mNew = mOld;
#pragma unroll 8
      for (int c2 = 0; c2 < 64; ++c2) mNew = fmaxf(mNew, Sl[w][lane][c2]);
      float alpha = __expf(mOld - mNew);
      float sum = 0.0f;
#pragma unroll 8
      for (int c2 = 0; c2 < 64; ++c2) {
        float p = __expf(Sl[w][lane][c2] - mNew);
        sum += p;
        Pl[w][lane][c2] = (__bf16)p;
      }
      Ml[w][lane] = mNew;
      Ll[w][lane] = Ll[w][lane] * alpha + sum;
      Al[w][lane] = alpha;
    }
    __syncthreads();

#pragma unroll
    for (int j = 0; j < 8; ++j)
#pragma unroll
      for (int vr = 0; vr < 8; ++vr) o[j][vr] *= Al[w][mg + vr];

    // O += P * V
#pragma unroll
    for (int kk = 0; kk < 2; ++kk) {
      v16bf pf = ldfrag(&Pl[w][r][kk * 32 + koff]);
#pragma unroll
      for (int j = 0; j < 8; ++j) {
        v16bf vf = ldfrag(&Vt[j * 16 + r][kk * 32 + koff]);
        o[j] = wmma_bf16(pf, vf, o[j]);
      }
    }
  }

  __syncthreads();
#pragma unroll
  for (int j = 0; j < 8; ++j)
#pragma unroll
    for (int vr = 0; vr < 8; ++vr) {
      int m = mg + vr;
      float inv = 1.0f / Ll[w][m];
      int row = q0 + w * 16 + m;
      Ctx[(size_t)row * NSTATE + h * HDIM + j * 16 + r] =
          (__bf16)(o[j][vr] * inv);
    }
}

// ---------------------------------------------------------------------------
// Launch
// ---------------------------------------------------------------------------
extern "C" void kernel_launch(void* const* d_in, const int* in_sizes, int n_in,
                              void* d_out, int out_size, void* d_ws,
                              size_t ws_size, hipStream_t stream) {
  const float* x    = (const float*)d_in[0];
  const float* cosT = (const float*)d_in[1];
  const float* sinT = (const float*)d_in[2];
  // d_in[3] attention_mask: causal mask recomputed inline
  const float* kc   = (const float*)d_in[4];
  const float* vc   = (const float*)d_in[5];
  const float* wq   = (const float*)d_in[6];
  const float* wk   = (const float*)d_in[7];
  const float* wv   = (const float*)d_in[8];
  const float* wo   = (const float*)d_in[9];

  float* out  = (float*)d_out;
  float* kout = out + (size_t)SEQ * NSTATE;
  float* vout = kout + (size_t)TOTAL * NSTATE;

  __bf16* xb  = (__bf16*)d_ws;
  __bf16* qrw = xb  + (size_t)SEQ   * NSTATE;
  __bf16* kb  = qrw + (size_t)SEQ   * NSTATE;
  __bf16* vb  = kb  + (size_t)TOTAL * NSTATE;
  __bf16* ctx = vb  + (size_t)TOTAL * NSTATE;

  const int threads = 256;

  k_cvt_bf16<<<(SEQ * NSTATE / 4 + threads - 1) / threads, threads, 0, stream>>>(
      x, xb, SEQ * NSTATE);

  dim3 gg(NSTATE / 128, SEQ / 128);
  k_gemm_nt<true><<<gg, 256, 0, stream>>>(xb, wq, qrw, NSTATE, NSTATE);
  k_gemm_nt<true><<<gg, 256, 0, stream>>>(xb, wk, kb + (size_t)PAST * NSTATE,
                                          NSTATE, NSTATE);
  k_gemm_nt<true><<<gg, 256, 0, stream>>>(xb, wv, vb + (size_t)PAST * NSTATE,
                                          NSTATE, NSTATE);

  int nrope = SEQ * NHEAD * 64;
  k_rope_q<<<(nrope + threads - 1) / threads, threads, 0, stream>>>(qrw, cosT,
                                                                    sinT);
  k_rope_k<<<(nrope + threads - 1) / threads, threads, 0, stream>>>(kb, kout,
                                                                    cosT, sinT);

  int ncache = PAST * NSTATE / 4;
  k_cache_store<<<(ncache + threads - 1) / threads, threads, 0, stream>>>(
      kc, kb, kout);
  k_cache_store<<<(ncache + threads - 1) / threads, threads, 0, stream>>>(
      vc, vb, vout);

  int nv = SEQ * NSTATE / 4;
  k_vnew_store<<<(nv + threads - 1) / threads, threads, 0, stream>>>(vb, vout);

  k_attn<<<dim3(SEQ / 64, NHEAD), 128, 0, stream>>>(qrw, kb, vb, ctx);

  k_gemm_nt<false><<<gg, 256, 0, stream>>>(ctx, wo, out, NSTATE, NSTATE);
}
